// SRU_Formula_Cell_41403484734021
// MI455X (gfx1250) — matrix operand order repaired and checked by reading, not compiled, not run on hardware
//
#include <hip/hip_runtime.h>

// ---------------------------------------------------------------------------
// SRU cell: 4 fused GEMMs (xt @ W^T for W_x/W_f/W_r/W_c) + sigmoid/tanh epilogue.
// fp32 in -> bf16 WMMA (v_wmma_f32_16x16x32_bf16), fp32 accumulation.
//
// Data movement (CDNA5 async path):
//   global fp32 --global_load_async_to_lds_b128--> LDS stage (48 KB, ASYNCcnt)
//   LDS stage --convert pass (pack bf16)--> double-buffered bf16 tiles (60 KB)
//   bf16 tiles --ds_load_b128 frags--> v_wmma_f32_16x16x32_bf16 (16/wave/iter)
// DMA for tile t+1 is in flight while tile t is computed; all A+B fragments are
// preloaded so the 16 WMMAs issue back-to-back behind one DS wait.
// Workgroup tile M=128 x N=64, K-step 32; 8 waves in 4(M) x 2(N); each wave
// owns 32x32 per gate -> 16 v8f accumulators.
// ---------------------------------------------------------------------------

typedef __attribute__((ext_vector_type(16))) __bf16 v16bf;
typedef __attribute__((ext_vector_type(8)))  float  v8f;

#define B_SZ 8192
#define K_SZ 2048
#define N_SZ 2048
#define BM 128
#define BN 64
#define BK 32
#define LDK 40                  // 32 + 8 pad: 80B rows, conflict-free frag reads
#define NIT (K_SZ / BK)         // 64 K-steps

#define BFA_BYTES (2 * BM * LDK * 2)            // 20480
#define BFB_BYTES (2 * 4 * BN * LDK * 2)        // 40960
#define STG_FLOATS (BM * BK + 4 * BN * BK)      // 12288 (A then B)
#define LDS_BYTES (BFA_BYTES + BFB_BYTES + STG_FLOATS * 4)  // 110592

union FragBF {
    v16bf v;
    uint4 q[2];
};

// pack two fp32 -> packed bf16 pair (RNE), pure 32-bit VALU ops
__device__ __forceinline__ unsigned int pack2bf(float lo, float hi) {
    union { float f; unsigned int u; } a, b;
    a.f = lo; b.f = hi;
    const unsigned int ra = (a.u + 0x7FFFu + ((a.u >> 16) & 1u)) >> 16;
    const unsigned int rb = (b.u + 0x7FFFu + ((b.u >> 16) & 1u)) & 0xFFFF0000u;
    return ra | rb;
}

__device__ __forceinline__ uint2 pack4bf(float4 v) {
    return make_uint2(pack2bf(v.x, v.y), pack2bf(v.z, v.w));
}

// fire-and-forget 16B async copy: global (SGPR base + VGPR offset) -> LDS
__device__ __forceinline__ void async_b128(unsigned lds_off, int voff,
                                           unsigned long long sbase) {
    asm volatile("global_load_async_to_lds_b128 %0, %1, %2"
                 :: "v"(lds_off), "v"(voff), "s"(sbase) : "memory");
}

__device__ __forceinline__ void wait_async0() {
    asm volatile("s_wait_asynccnt 0x0" ::: "memory");
}

__global__ __launch_bounds__(256)
void sru_fused_kernel(const float* __restrict__ xt,
                      const float* __restrict__ ctf,
                      const float* __restrict__ Wx,
                      const float* __restrict__ Wf,
                      const float* __restrict__ bf,
                      const float* __restrict__ Wr,
                      const float* __restrict__ br,
                      const float* __restrict__ Wc,
                      const float* __restrict__ bc,
                      float* __restrict__ out_ht,
                      float* __restrict__ out_ct)
{
    extern __shared__ __align__(16) char smem[];
    unsigned short (*bfA)[BM * LDK] =
        reinterpret_cast<unsigned short(*)[BM * LDK]>(smem);
    unsigned short (*bfB)[4][BN * LDK] =
        reinterpret_cast<unsigned short(*)[4][BN * LDK]>(smem + BFA_BYTES);
    float* stg = reinterpret_cast<float*>(smem + BFA_BYTES + BFB_BYTES);

    const int tid  = threadIdx.x;
    const int wave = tid >> 5;
    const int lane = tid & 31;
    const int l    = lane & 15;
    const int hi   = lane >> 4;

    const int blockN = blockIdx.x * BN;
    const int blockM = blockIdx.y * BM;

    const int waveM = (wave & 3) * 32;   // 4 waves along M
    const int waveN = (wave >> 2) * 32;  // 2 waves along N

    v8f acc[4][2][2];
#pragma unroll
    for (int g = 0; g < 4; g++)
#pragma unroll
        for (int m = 0; m < 2; m++)
#pragma unroll
            for (int n = 0; n < 2; n++)
                acc[g][m][n] = (v8f){0.f, 0.f, 0.f, 0.f, 0.f, 0.f, 0.f, 0.f};

    // ---- async DMA issue for K-block k0: 12 x b128 per thread -------------
    // flat float4 index: [0,1024) = A tile (128x32), [1024,3072) = B tiles
    auto asyncLoad = [&](int k0) {
#pragma unroll
        for (int i = 0; i < 12; i++) {
            const int idx = tid + i * 256;
            if (i < 4) {                           // A region
                const int row = idx >> 3;          // 8 float4 per row
                const int c4  = idx & 7;
                const unsigned lds = (unsigned)(size_t)&stg[idx * 4];
                const int voff = ((blockM + row) * K_SZ + c4 * 4) * 4;
                async_b128(lds, voff, (unsigned long long)(size_t)(xt + k0));
            } else {                               // B region; g compile-time
                const int idxB = idx - 1024;
                const int g    = (i - 4) >> 1;
                const float* __restrict__ W =
                    (g == 0) ? Wx : (g == 1) ? Wf : (g == 2) ? Wr : Wc;
                const int rem = idxB & 511;
                const int row = rem >> 3;
                const int c4  = rem & 7;
                const unsigned lds = (unsigned)(size_t)&stg[BM * BK + idxB * 4];
                const int voff = ((blockN + row) * K_SZ + c4 * 4) * 4;
                async_b128(lds, voff, (unsigned long long)(size_t)(W + k0));
            }
        }
    };

    // ---- convert pass: fp32 stage -> bf16 tile buffer ---------------------
    auto cvt = [&](int buf) {
#pragma unroll
        for (int i = 0; i < 12; i++) {
            const int idx = tid + i * 256;
            if (i < 4) {
                const float4 v = *(const float4*)&stg[idx * 4];
                const int row = idx >> 3;
                const int c4  = idx & 7;
                *(uint2*)&bfA[buf][row * LDK + c4 * 4] = pack4bf(v);
            } else {
                const int idxB = idx - 1024;
                const float4 v = *(const float4*)&stg[BM * BK + idxB * 4];
                const int g   = (i - 4) >> 1;
                const int rem = idxB & 511;
                const int row = rem >> 3;
                const int c4  = rem & 7;
                *(uint2*)&bfB[buf][g][row * LDK + c4 * 4] = pack4bf(v);
            }
        }
    };

    // ---- 16 WMMAs on one bf16 buffer --------------------------------------
    auto compute = [&](int buf) {
        // A frags: lanes 0-15 K={0..7,16..23}, lanes 16-31 K={8..15,24..31}
        FragBF a[2];
#pragma unroll
        for (int m = 0; m < 2; m++) {
            const unsigned short* p = &bfA[buf][(waveM + m * 16 + l) * LDK + hi * 8];
            a[m].q[0] = *(const uint4*)(p);
            a[m].q[1] = *(const uint4*)(p + 16);
        }
        // preload ALL B frags (8 x 8 VGPRs) so the 16 WMMAs below issue
        // back-to-back behind a single DS wait instead of 8 full drains
        FragBF b[4][2];
#pragma unroll
        for (int g = 0; g < 4; g++)
#pragma unroll
            for (int n = 0; n < 2; n++) {
                const unsigned short* p = &bfB[buf][g][(waveN + n * 16 + l) * LDK + hi * 8];
                b[g][n].q[0] = *(const uint4*)(p);
                b[g][n].q[1] = *(const uint4*)(p + 16);
            }
#pragma unroll
        for (int g = 0; g < 4; g++)
#pragma unroll
            for (int n = 0; n < 2; n++)
#pragma unroll
                for (int m = 0; m < 2; m++)
                    acc[g][m][n] = __builtin_amdgcn_wmma_f32_16x16x32_bf16(
                        false, a[m].v, false, b[g][n].v, (short)0, acc[g][m][n],
                        false, false);
    };

    // ---- async software pipeline ------------------------------------------
    asyncLoad(0);                 // tile 0 -> stage
    wait_async0();
    __syncthreads();              // every wave waited own ASYNCcnt -> tile 0 visible
    cvt(0);                       // stage -> bf[0]
    __syncthreads();              // bf[0] ready; stage free
    asyncLoad(BK);                // tile 1 DMA in flight

#pragma unroll 1
    for (int it = 0; it < NIT; ++it) {
        compute(it & 1);                      // WMMA tile it; DMA overlaps
        if (it + 1 < NIT) {
            wait_async0();                    // own slice of tile it+1 arrived
            __syncthreads();                  // all slices arrived; compute(it) done
            cvt((it + 1) & 1);                // stage -> bf[nxt]
            __syncthreads();                  // bf[nxt] ready; stage free
            if (it + 2 < NIT) asyncLoad((it + 2) * BK);
        }
    }

    // ---- fused SRU epilogue ----------------------------------------------
    // C/D layout: VGPR j = (row = base + j + 8*hi, col = base + (lane&15))
#pragma unroll
    for (int n = 0; n < 2; n++) {
        const int col  = blockN + waveN + n * 16 + l;
        const float bfv = bf[col];
        const float brv = br[col];
        const float bcv = bc[col];
#pragma unroll
        for (int m = 0; m < 2; m++) {
            const int rowBase = blockM + waveM + m * 16 + hi * 8;
#pragma unroll
            for (int j = 0; j < 8; j++) {
                const size_t off = (size_t)(rowBase + j) * N_SZ + col;
                const float x_t  = acc[0][m][n][j];
                const float fpre = acc[1][m][n][j] + bfv;
                const float rpre = acc[2][m][n][j] + brv;
                const float xcv  = acc[3][m][n][j] + bcv;
                const float ft = 1.0f / (1.0f + __expf(-fpre));
                const float rt = 1.0f / (1.0f + __expf(-rpre));
                const float ct = ft * ctf[off] + (1.0f - ft) * x_t;
                const float th = 1.0f - 2.0f / (__expf(2.0f * ct) + 1.0f);  // tanh
                const float ht = rt * th + (1.0f - rt) * xcv;
                out_ht[off] = ht;
                out_ct[off] = ct;
            }
        }
    }
}

extern "C" void kernel_launch(void* const* d_in, const int* in_sizes, int n_in,
                              void* d_out, int out_size, void* d_ws, size_t ws_size,
                              hipStream_t stream) {
    (void)in_sizes; (void)n_in; (void)out_size; (void)d_ws; (void)ws_size;
    const float* xt  = (const float*)d_in[0];
    const float* ctf = (const float*)d_in[1];
    const float* Wx  = (const float*)d_in[2];
    const float* Wf  = (const float*)d_in[3];
    const float* bf  = (const float*)d_in[4];
    const float* Wr  = (const float*)d_in[5];
    const float* br  = (const float*)d_in[6];
    const float* Wc  = (const float*)d_in[7];
    const float* bc  = (const float*)d_in[8];

    float* out_ht = (float*)d_out;
    float* out_ct = out_ht + (size_t)B_SZ * N_SZ;

    static_assert(LDS_BYTES == 110592, "LDS budget");
    (void)hipFuncSetAttribute(reinterpret_cast<const void*>(sru_fused_kernel),
                              hipFuncAttributeMaxDynamicSharedMemorySize,
                              LDS_BYTES);

    dim3 grid(N_SZ / BN, B_SZ / BM);  // 32 x 64 = 2048 workgroups
    dim3 block(256);
    sru_fused_kernel<<<grid, block, LDS_BYTES, stream>>>(xt, ctf, Wx, Wf, bf, Wr, br,
                                                         Wc, bc, out_ht, out_ct);
}
